// GAUAlpha_48627619726043
// MI455X (gfx1250) — matrix operand order repaired and checked by reading, not compiled, not run on hardware
//
#include <hip/hip_runtime.h>
#include <hip/hip_bf16.h>

// ---------------------------------------------------------------------------
// GAU-alpha block for MI455X (gfx1250, wave32, WMMA).
// All GEMMs use v_wmma_f32_16x16x32_f16 (f16 in, f32 accumulate).
// Block tile: 128x128, K-step 32, 256 threads = 8 waves, wave tile 32x64.
// Double-buffered LDS pipeline; async global->LDS copies when available.
// ---------------------------------------------------------------------------

typedef __attribute__((ext_vector_type(16))) _Float16 v16h;
typedef __attribute__((ext_vector_type(8)))  float    v8f;
typedef __attribute__((ext_vector_type(4)))  int      v4i;

#define B_DIM 64
#define S_SEQ 512
#define H_DIM 512
#define E_DIM 1024
#define SD    128                    // S_DIM
#define UVDIM (2 * E_DIM + SD)       // 2176
#define NROWS ((long)B_DIM * S_SEQ)  // 32768
#define TBUF  (128 * 32)             // halves per LDS tile buffer

#if __has_builtin(__builtin_amdgcn_global_load_async_to_lds_b128) && \
    __has_builtin(__builtin_amdgcn_s_wait_asynccnt)
#define USE_ASYNC 1
#define WAIT_ASYNC(n) __builtin_amdgcn_s_wait_asynccnt(n)
#else
#define USE_ASYNC 0
#define WAIT_ASYNC(n)
#endif

// 16-byte global -> LDS copy (async engine when available)
__device__ __forceinline__ void cp16(const _Float16* g, _Float16* l) {
#if USE_ASYNC
  typedef __attribute__((address_space(1))) v4i gv4i;
  typedef __attribute__((address_space(3))) v4i lv4i;
  __builtin_amdgcn_global_load_async_to_lds_b128(
      (gv4i*)(__attribute__((address_space(1))) void*)const_cast<_Float16*>(g),
      (lv4i*)(__attribute__((address_space(3))) void*)l, 0, 0);
#else
  *(uint4*)l = *(const uint4*)g;
#endif
}

// ------------------------------- WMMA core ---------------------------------
// LDS tiles: As[128][32] halves (A, row-major M x K slice)
//            Bs[128][32] halves (B stored n-major: Bs[n][k])
//
// Fragment layouts per CDNA5 ISA 7.12.2 (wave32):
//  A 16x32 f16 : lane L holds row M=L%16; VGPR v<4: K=2v,2v+1 (+8 if L>=16),
//                VGPR v>=4: K=16+2(v-4),+1 (+8 if L>=16)
//  B 32x16 f16 : lane L holds col N=L%16; this code (A too) uses K = (L<16?0:16) + element index
//  C/D 16x16 f32: lane L, reg r -> M = r + (L>=16?8:0), N = L%16

__device__ __forceinline__ void mma_step(const _Float16* As, const _Float16* Bs,
                                         int wm, int wn, int lane,
                                         v8f acc[2][4]) {
  const int hi = (lane >> 4) & 1;
  const int lr = lane & 15;
  v16h af[2];
  v16h bf[4];
#pragma unroll
  for (int i = 0; i < 2; ++i) {
    const _Float16* row = As + (wm + i * 16 + lr) * 32;
    const int b = hi ? 16 : 0;
#pragma unroll
    for (int e = 0; e < 8; ++e) {
      af[i][e]     = row[b + e];        // K = b .. b+7
      af[i][e + 8] = row[b + 8 + e];    // K = b+8 .. b+15
    }
  }
#pragma unroll
  for (int j = 0; j < 4; ++j) {
    const _Float16* row = Bs + (wn + j * 16 + lr) * 32;
    const int b = hi ? 16 : 0;
#pragma unroll
    for (int e = 0; e < 16; ++e) bf[j][e] = row[b + e];
  }
#pragma unroll
  for (int i = 0; i < 2; ++i)
#pragma unroll
    for (int j = 0; j < 4; ++j)
      acc[i][j] = __builtin_amdgcn_wmma_f32_16x16x32_f16(
          false, af[i], false, bf[j], (short)0, acc[i][j], false, false);
}

// B source is row-major N x K (weight-style): Bglobal[n][k]
// As/Bs are double buffers of 2*TBUF halves each.
__device__ __forceinline__ void gemm_nk(const _Float16* __restrict__ A, long lda, long arow0,
                                        const _Float16* __restrict__ B, long ldb, long brow0,
                                        int K, _Float16* As, _Float16* Bs,
                                        v8f acc[2][4]) {
  const int t    = threadIdx.x;
  const int wave = t >> 5, lane = t & 31;
  const int wm   = (wave & 3) * 32, wn = (wave >> 2) * 64;
  const int lr   = t >> 1;          // 0..127  tile row
  const int lc   = (t & 1) * 16;    // 0 or 16 halves within K-step
  const _Float16* ap = A + (arow0 + lr) * lda + lc;
  const _Float16* bp = B + (brow0 + lr) * ldb + lc;
  _Float16* asl = As + lr * 32 + lc;
  _Float16* bsl = Bs + lr * 32 + lc;
  const int nk = K >> 5;
  int c = 0;
  // prologue: tile 0 -> buffer 0 (4 wave-wide copies)
  cp16(ap, asl); cp16(ap + 8, asl + 8);
  cp16(bp, bsl); cp16(bp + 8, bsl + 8);
  for (int kk = 0; kk < nk; ++kk) {
    if (kk + 1 < nk) {   // issue next tile into the other buffer, then wait tile kk
      _Float16* an = asl + (c ^ 1) * TBUF;
      _Float16* bn = bsl + (c ^ 1) * TBUF;
      cp16(ap + 32, an); cp16(ap + 40, an + 8);
      cp16(bp + 32, bn); cp16(bp + 40, bn + 8);
      __builtin_prefetch(ap + 64, 0, 3);
      __builtin_prefetch(bp + 64, 0, 3);
      WAIT_ASYNC(4);
    } else {
      WAIT_ASYNC(0);
    }
    __syncthreads();
    mma_step(As + c * TBUF, Bs + c * TBUF, wm, wn, lane, acc);
    __syncthreads();
    ap += 32; bp += 32;
    c ^= 1;
  }
}

// B source is row-major K x N (activation-style): Bglobal[k][n]; transpose on LDS store
__device__ __forceinline__ void gemm_kn(const _Float16* __restrict__ A, long lda, long arow0,
                                        const _Float16* __restrict__ B, long ldb, long bcol0,
                                        int K, _Float16* As, _Float16* Bs,
                                        v8f acc[2][4]) {
  const int t    = threadIdx.x;
  const int wave = t >> 5, lane = t & 31;
  const int wm   = (wave & 3) * 32, wn = (wave >> 2) * 64;
  const int lr   = t >> 1;
  const int lc   = (t & 1) * 16;
  const int bk   = t & 31;           // k within K-step
  const int bn   = (t >> 5) * 16;    // 16-col group within N tile
  const _Float16* ap = A + (arow0 + lr) * lda + lc;
  const _Float16* bp = B + (long)bk * ldb + bcol0 + bn;
  _Float16* asl = As + lr * 32 + lc;
  const int nk = K >> 5;
  int c = 0;
  // prologue tile 0
  cp16(ap, asl); cp16(ap + 8, asl + 8);
  {
    union { uint4 q[2]; _Float16 h[16]; } tb;
    tb.q[0] = *(const uint4*)(bp);
    tb.q[1] = *(const uint4*)(bp + 8);
#pragma unroll
    for (int i = 0; i < 16; ++i) Bs[(bn + i) * 32 + bk] = tb.h[i];
  }
  for (int kk = 0; kk < nk; ++kk) {
    if (kk + 1 < nk) {
      _Float16* an = asl + (c ^ 1) * TBUF;
      cp16(ap + 32, an); cp16(ap + 40, an + 8);
      union { uint4 q[2]; _Float16 h[16]; } tb;
      tb.q[0] = *(const uint4*)(bp + 32 * ldb);
      tb.q[1] = *(const uint4*)(bp + 32 * ldb + 8);
      _Float16* bsb = Bs + (c ^ 1) * TBUF;
#pragma unroll
      for (int i = 0; i < 16; ++i) bsb[(bn + i) * 32 + bk] = tb.h[i];
      __builtin_prefetch(ap + 64, 0, 3);
      __builtin_prefetch(bp + 64 * ldb, 0, 3);
      WAIT_ASYNC(2);
    } else {
      WAIT_ASYNC(0);
    }
    __syncthreads();
    mma_step(As + c * TBUF, Bs + c * TBUF, wm, wn, lane, acc);
    __syncthreads();
    ap += 32; bp += 32 * ldb;
    c ^= 1;
  }
}

#define EPILOGUE_VARS                                     \
  const int t_    = threadIdx.x;                          \
  const int wave_ = t_ >> 5, lane_ = t_ & 31;             \
  const int wm_ = (wave_ & 3) * 32, wn_ = (wave_ >> 2) * 64; \
  const int hi_ = (lane_ >> 4) & 1, lr_ = lane_ & 15;

// ------------------------------- kernels -----------------------------------

__global__ __launch_bounds__(256) void cvt_f16_kernel(const float* __restrict__ src,
                                                      _Float16* __restrict__ dst, int n) {
  int i = blockIdx.x * 256 + threadIdx.x;
  if (i < n) dst[i] = (_Float16)src[i];
}

// one wave per row of 512
__global__ __launch_bounds__(256) void rmsnorm_kernel(const float* __restrict__ x,
                                                      const float* __restrict__ ln_g,
                                                      _Float16* __restrict__ xn) {
  const int wave = threadIdx.x >> 5, lane = threadIdx.x & 31;
  const long row = (long)blockIdx.x * 8 + wave;
  const float* xr = x + row * H_DIM;
  float vals[16];
  float s = 0.f;
#pragma unroll
  for (int i = 0; i < 16; ++i) { vals[i] = xr[lane + i * 32]; s += vals[i] * vals[i]; }
#pragma unroll
  for (int off = 16; off > 0; off >>= 1) s += __shfl_xor(s, off, 32);
  const float norm  = sqrtf(s * (1.0f / H_DIM));
  const float scale = ln_g[0] * __builtin_amdgcn_rcpf(fmaxf(norm, 1e-5f));
  _Float16* xo = xn + row * H_DIM;
#pragma unroll
  for (int i = 0; i < 16; ++i) xo[lane + i * 32] = (_Float16)(vals[i] * scale);
}

// act = silu(xn @ uv_w^T + uv_b), split into u / v / base
__global__ __launch_bounds__(256) void uv_gemm_kernel(const _Float16* __restrict__ A,
                                                      const _Float16* __restrict__ W,
                                                      const float* __restrict__ uv_b,
                                                      _Float16* __restrict__ u,
                                                      _Float16* __restrict__ v,
                                                      _Float16* __restrict__ base) {
  __shared__ _Float16 As[2 * TBUF];
  __shared__ _Float16 Bs[2 * TBUF];
  const long m0 = (long)blockIdx.x * 128;
  const long n0 = (long)blockIdx.y * 128;
  v8f acc[2][4];
#pragma unroll
  for (int i = 0; i < 2; ++i)
#pragma unroll
    for (int j = 0; j < 4; ++j) acc[i][j] = 0.f;
  gemm_nk(A, H_DIM, m0, W, H_DIM, n0, H_DIM, As, Bs, acc);
  EPILOGUE_VARS
#pragma unroll
  for (int i = 0; i < 2; ++i)
#pragma unroll
    for (int j = 0; j < 4; ++j)
#pragma unroll
      for (int r = 0; r < 8; ++r) {
        const long row = m0 + wm_ + i * 16 + hi_ * 8 + r;
        const int  col = (int)n0 + wn_ + j * 16 + lr_;
        float c = acc[i][j][r] + uv_b[col];
        float a = c * __builtin_amdgcn_rcpf(1.f + __expf(-c));  // silu
        if (col < E_DIM)            u[row * E_DIM + col]               = (_Float16)a;
        else if (col < 2 * E_DIM)   v[row * E_DIM + (col - E_DIM)]     = (_Float16)a;
        else                        base[row * SD + (col - 2 * E_DIM)] = (_Float16)a;
      }
}

// base -> gamma/beta -> rope -> q,k
__global__ __launch_bounds__(256) void rope_kernel(const _Float16* __restrict__ base,
                                                   const float* __restrict__ gamma,
                                                   const float* __restrict__ beta,
                                                   _Float16* __restrict__ q,
                                                   _Float16* __restrict__ k) {
  const int  idx = blockIdx.x * 256 + threadIdx.x;   // row*64 + d
  const long row = idx >> 6;
  const int  d   = idx & 63;
  const int  pos = (int)(row & (S_SEQ - 1));
  const float g1 = (float)base[row * SD + d];
  const float g2 = (float)base[row * SD + d + 64];
  // inv_freq = 10000^(d/64) = exp2(d * log2(10000)/64)
  const float invf = exp2f((float)d * (13.287712379549449f / 64.f));
  float sn, cs;
  sincosf((float)pos * invf, &sn, &cs);
#pragma unroll
  for (int j = 0; j < 2; ++j) {
    const float b1 = g1 * gamma[j * SD + d]      + beta[j * SD + d];
    const float b2 = g2 * gamma[j * SD + d + 64] + beta[j * SD + d + 64];
    _Float16* dst = j ? k : q;
    dst[row * SD + d]      = (_Float16)(b1 * cs - b2 * sn);
    dst[row * SD + d + 64] = (_Float16)(b2 * cs + b1 * sn);
  }
}

// scores = relu((q k^T + bias)/sqrt(128))^2 per batch
__global__ __launch_bounds__(256) void qk_kernel(const _Float16* __restrict__ q,
                                                 const _Float16* __restrict__ kmat,
                                                 const float* __restrict__ w_bias,
                                                 _Float16* __restrict__ scores) {
  __shared__ _Float16 As[2 * TBUF];
  __shared__ _Float16 Bs[2 * TBUF];
  const int  b  = blockIdx.z;
  const long m0 = (long)blockIdx.x * 128;
  const long n0 = (long)blockIdx.y * 128;
  const long boff = (long)b * S_SEQ * SD;
  v8f acc[2][4];
#pragma unroll
  for (int i = 0; i < 2; ++i)
#pragma unroll
    for (int j = 0; j < 4; ++j) acc[i][j] = 0.f;
  gemm_nk(q + boff, SD, m0, kmat + boff, SD, n0, SD, As, Bs, acc);
  EPILOGUE_VARS
  const float inv_sqrt_s = 0.088388347648318447f;  // 1/sqrt(128)
  _Float16* so = scores + (long)b * S_SEQ * S_SEQ;
#pragma unroll
  for (int i = 0; i < 2; ++i)
#pragma unroll
    for (int j = 0; j < 4; ++j)
#pragma unroll
      for (int r = 0; r < 8; ++r) {
        const int srow = (int)m0 + wm_ + i * 16 + hi_ * 8 + r;
        const int tcol = (int)n0 + wn_ + j * 16 + lr_;
        float c  = acc[i][j][r] + w_bias[tcol - srow + (S_SEQ - 1)];
        float rl = fmaxf(c * inv_sqrt_s, 0.f);
        so[(long)srow * S_SEQ + tcol] = (_Float16)(rl * rl);
      }
}

// y = u * (scores @ v) per batch
__global__ __launch_bounds__(256) void av_kernel(const _Float16* __restrict__ scores,
                                                 const _Float16* __restrict__ v,
                                                 const _Float16* __restrict__ u,
                                                 _Float16* __restrict__ y) {
  __shared__ _Float16 As[2 * TBUF];
  __shared__ _Float16 Bs[2 * TBUF];
  const int  b  = blockIdx.z;
  const long m0 = (long)blockIdx.x * 128;
  const long n0 = (long)blockIdx.y * 128;
  v8f acc[2][4];
#pragma unroll
  for (int i = 0; i < 2; ++i)
#pragma unroll
    for (int j = 0; j < 4; ++j) acc[i][j] = 0.f;
  gemm_kn(scores + (long)b * S_SEQ * S_SEQ, S_SEQ, m0,
          v + (long)b * S_SEQ * E_DIM, E_DIM, n0, S_SEQ, As, Bs, acc);
  EPILOGUE_VARS
#pragma unroll
  for (int i = 0; i < 2; ++i)
#pragma unroll
    for (int j = 0; j < 4; ++j)
#pragma unroll
      for (int r = 0; r < 8; ++r) {
        const long grow = (long)b * S_SEQ + m0 + wm_ + i * 16 + hi_ * 8 + r;
        const int  col  = (int)n0 + wn_ + j * 16 + lr_;
        const float uu  = (float)u[grow * E_DIM + col];
        y[grow * E_DIM + col] = (_Float16)(acc[i][j][r] * uu);
      }
}

// out = y @ o_w^T + o_b + x
__global__ __launch_bounds__(256) void out_gemm_kernel(const _Float16* __restrict__ y,
                                                       const _Float16* __restrict__ ow,
                                                       const float* __restrict__ o_b,
                                                       const float* __restrict__ x,
                                                       float* __restrict__ out) {
  __shared__ _Float16 As[2 * TBUF];
  __shared__ _Float16 Bs[2 * TBUF];
  const long m0 = (long)blockIdx.x * 128;
  const long n0 = (long)blockIdx.y * 128;
  v8f acc[2][4];
#pragma unroll
  for (int i = 0; i < 2; ++i)
#pragma unroll
    for (int j = 0; j < 4; ++j) acc[i][j] = 0.f;
  gemm_nk(y, E_DIM, m0, ow, E_DIM, n0, E_DIM, As, Bs, acc);
  EPILOGUE_VARS
#pragma unroll
  for (int i = 0; i < 2; ++i)
#pragma unroll
    for (int j = 0; j < 4; ++j)
#pragma unroll
      for (int r = 0; r < 8; ++r) {
        const long row = m0 + wm_ + i * 16 + hi_ * 8 + r;
        const int  col = (int)n0 + wn_ + j * 16 + lr_;
        out[row * H_DIM + col] = acc[i][j][r] + o_b[col] + x[row * H_DIM + col];
      }
}

// ------------------------------- launcher ----------------------------------

extern "C" void kernel_launch(void* const* d_in, const int* in_sizes, int n_in,
                              void* d_out, int out_size, void* d_ws, size_t ws_size,
                              hipStream_t stream) {
  const float* x      = (const float*)d_in[0];
  const float* ln_g   = (const float*)d_in[1];
  const float* uv_w   = (const float*)d_in[2];
  const float* uv_b   = (const float*)d_in[3];
  const float* gamma  = (const float*)d_in[4];
  const float* beta   = (const float*)d_in[5];
  const float* w_bias = (const float*)d_in[6];
  const float* o_w    = (const float*)d_in[7];
  const float* o_b    = (const float*)d_in[8];
  float* out = (float*)d_out;

  char* ws = (char*)d_ws;
  auto alloc_h = [&ws](size_t halves) {
    _Float16* p = (_Float16*)ws;
    ws += ((halves * sizeof(_Float16) + 255) & ~(size_t)255);
    return p;
  };
  _Float16* uv_w_h = alloc_h((size_t)UVDIM * H_DIM);
  _Float16* o_w_h  = alloc_h((size_t)H_DIM * E_DIM);
  _Float16* xn     = alloc_h((size_t)NROWS * H_DIM);
  _Float16* u      = alloc_h((size_t)NROWS * E_DIM);
  _Float16* v      = alloc_h((size_t)NROWS * E_DIM);
  _Float16* baseb  = alloc_h((size_t)NROWS * SD);
  _Float16* qh     = alloc_h((size_t)NROWS * SD);
  _Float16* kh     = alloc_h((size_t)NROWS * SD);
  _Float16* sc     = alloc_h((size_t)B_DIM * S_SEQ * S_SEQ);
  _Float16* yh     = alloc_h((size_t)NROWS * E_DIM);

  const int n_uvw = UVDIM * H_DIM;
  const int n_ow  = H_DIM * E_DIM;
  cvt_f16_kernel<<<(n_uvw + 255) / 256, 256, 0, stream>>>(uv_w, uv_w_h, n_uvw);
  cvt_f16_kernel<<<(n_ow + 255) / 256, 256, 0, stream>>>(o_w, o_w_h, n_ow);

  rmsnorm_kernel<<<NROWS / 8, 256, 0, stream>>>(x, ln_g, xn);

  dim3 g_uv(NROWS / 128, UVDIM / 128);          // 256 x 17
  uv_gemm_kernel<<<g_uv, 256, 0, stream>>>(xn, uv_w_h, uv_b, u, v, baseb);

  rope_kernel<<<(int)(NROWS * 64 / 256), 256, 0, stream>>>(baseb, gamma, beta, qh, kh);

  dim3 g_qk(S_SEQ / 128, S_SEQ / 128, B_DIM);   // 4 x 4 x 64
  qk_kernel<<<g_qk, 256, 0, stream>>>(qh, kh, w_bias, sc);

  dim3 g_av(S_SEQ / 128, E_DIM / 128, B_DIM);   // 4 x 8 x 64
  av_kernel<<<g_av, 256, 0, stream>>>(sc, v, u, yh);

  dim3 g_o(NROWS / 128, H_DIM / 128);           // 256 x 4
  out_gemm_kernel<<<g_o, 256, 0, stream>>>(yh, o_w_h, o_b, x, out);
}
